// Net_58402965291510
// MI455X (gfx1250) — compile-verified, matrix operand order checked by reading, not run on hardware
//
#include <hip/hip_runtime.h>
#include <hip/hip_bf16.h>

// Sizes from reference: B=128, L=512, F=144, H=512
#define BB 128
#define LL 512
#define FF 144
#define FP 160   // F padded to multiple of 32 (WMMA K step)
#define HH 512

typedef __attribute__((ext_vector_type(8)))  float  v8f;
typedef __attribute__((ext_vector_type(8)))  __bf16 v8bf;
typedef __attribute__((ext_vector_type(16))) __bf16 v16bf;

#define LOG2E 1.4426950408889634f

__device__ inline float fast_sigmoid(float x) {
    // exp2 saturation gives exact 0/1 tails -> branch-free v_exp_f32 is safe
    float e = __builtin_amdgcn_exp2f(-LOG2E * x);
    return __builtin_amdgcn_rcpf(1.0f + e);
}

__device__ inline float fast_tanh(float x) {
    float a = __builtin_fabsf(x);
    float t = __builtin_amdgcn_exp2f(-2.0f * LOG2E * a);   // t in (0,1]
    float r = (1.0f - t) * __builtin_amdgcn_rcpf(1.0f + t);
    return __builtin_copysignf(r, x);
}

__device__ inline unsigned short f32_to_bf16(float f) {
    unsigned u = __builtin_bit_cast(unsigned, f);
    u += 0x7FFFu + ((u >> 16) & 1u);   // round to nearest even
    return (unsigned short)(u >> 16);
}

__device__ inline v16bf load_frag(const unsigned short* p) {
    // Per-lane fragment: 8 contiguous bf16 at p, 8 more at p+16 (ISA 16-bit 16x32 layout)
    v8bf lo = *(const v8bf*)(p);
    v8bf hi = *(const v8bf*)(p + 16);
    return __builtin_shufflevector(lo, hi, 0,1,2,3,4,5,6,7,8,9,10,11,12,13,14,15);
}

// ---------------- prep kernels ----------------

// Wt_ig[n*512 + k] = bf16(W_ig[k*512 + n])   (262144 elems)
__global__ void conv_wig(const float* __restrict__ W, unsigned short* __restrict__ Wt) {
    int idx = blockIdx.x * 256 + threadIdx.x;
    if (idx >= HH * HH) return;
    int n = idx >> 9, k = idx & (HH - 1);
    Wt[idx] = f32_to_bf16(W[k * HH + n]);
}

// Wt_ci[n*160 + k] = bf16(W_ci[k*512 + n]) for k<144, else 0   (512*160 elems)
__global__ void conv_wci(const float* __restrict__ W, unsigned short* __restrict__ Wt) {
    int idx = blockIdx.x * 256 + threadIdx.x;
    if (idx >= HH * FP) return;
    int n = idx / FP, k = idx % FP;
    Wt[idx] = (k < FF) ? f32_to_bf16(W[k * HH + n]) : (unsigned short)0;
}

// obs_bf[row*160 + f] = bf16(obs[row*144 + f]) for f<144, else 0   (65536*160 elems)
__global__ void conv_obs(const float* __restrict__ obs, unsigned short* __restrict__ ob) {
    long idx = (long)blockIdx.x * 256 + threadIdx.x;
    if (idx >= (long)BB * LL * FP) return;
    int  f   = (int)(idx % FP);
    long row = idx / FP;
    ob[idx] = (f < FF) ? f32_to_bf16(obs[row * FF + f]) : (unsigned short)0;
}

// c = 0, h0 = 0, out = b_out   (65536 elems each)
__global__ void init_state(float* __restrict__ c, unsigned short* __restrict__ h0,
                           float* __restrict__ out, const float* __restrict__ b_out) {
    int idx = blockIdx.x * 256 + threadIdx.x;
    if (idx >= BB * HH) return;  // BB*HH == BB*LL == 65536
    c[idx] = 0.0f;
    h0[idx] = 0;
    out[idx] = b_out[0];
}

// ---------------- phase 1: ci = tanh(obs @ W_ci + b_ci) ----------------
// M = 65536 (B*L), N = 512, K = 160 (padded). One wave per 16x16 tile.
__global__ __launch_bounds__(256)
void ci_gemm(const unsigned short* __restrict__ obs_bf,   // [65536 x 160] bf16
             const unsigned short* __restrict__ Wt_ci,    // [512 x 160]  bf16 (row = n)
             const float* __restrict__ b_ci,
             float* __restrict__ ci) {                    // [65536 x 512] f32
    const int lane = threadIdx.x & 31;
    const int wid  = (blockIdx.x * blockDim.x + threadIdx.x) >> 5;  // 0..131071
    const int tn = wid & 31;          // 32 N tiles
    const int tm = wid >> 5;          // 4096 M tiles
    const int l15  = lane & 15;
    const int hi   = lane >> 4;       // 0/1
    const int koff = hi * 8;

    const unsigned short* pa = obs_bf + (long)(tm * 16 + l15) * FP + koff;
    const unsigned short* pb = Wt_ci  + (long)(tn * 16 + l15) * FP + koff;

    v8f acc = {};
#pragma unroll
    for (int k0 = 0; k0 < FP; k0 += 32) {
        v16bf a = load_frag(pa + k0);
        v16bf b = load_frag(pb + k0);
        acc = __builtin_amdgcn_wmma_f32_16x16x32_bf16(false, a, false, b,
                                                      (short)0, acc, false, false);
    }

    const int n = tn * 16 + l15;
    const float bias = b_ci[n];
#pragma unroll
    for (int r = 0; r < 8; ++r) {
        int m = r + hi * 8;
        long row = (long)tm * 16 + m;
        ci[row * HH + n] = fast_tanh(acc[r] + bias);
    }
}

// ---------------- per-step recurrence ----------------
// ig = sigmoid(h_{t-1} @ W_ig + b_ig);  c += ci_t * ig;  h_t = c;
// out[b,t] += dot(c_row, W_out)   (fused, atomic per row)
__global__ __launch_bounds__(128)
void step_kernel(const unsigned short* __restrict__ hsrc,  // [128 x 512] bf16
                 unsigned short*       __restrict__ hdst,  // [128 x 512] bf16
                 const unsigned short* __restrict__ Wt,    // [512 x 512] bf16 (row = n)
                 const float* __restrict__ b_ig,
                 const float* __restrict__ ci,             // [65536 x 512] f32
                 float* __restrict__ c,                    // [128 x 512] f32
                 const float* __restrict__ W_out,          // [512]
                 float* __restrict__ out,                  // [128 x 512] (b*L + t)
                 int t) {
    const int lane = threadIdx.x & 31;
    const int wid  = (blockIdx.x * blockDim.x + threadIdx.x) >> 5;  // 0..255
    const int tm = wid & 7;           // 8 M tiles over B=128
    const int tn = wid >> 3;          // 32 N tiles over H=512
    const int rowBase = tm * 16;
    const int colBase = tn * 16;
    const int l15  = lane & 15;
    const int hi   = lane >> 4;
    const int koff = hi * 8;

    const unsigned short* pa = hsrc + (rowBase + l15) * HH + koff;
    const unsigned short* pb = Wt   + (colBase + l15) * HH + koff;

    v8f acc = {};
#pragma unroll
    for (int k0 = 0; k0 < HH; k0 += 32) {
        v16bf a = load_frag(pa + k0);
        v16bf b = load_frag(pb + k0);
        acc = __builtin_amdgcn_wmma_f32_16x16x32_bf16(false, a, false, b,
                                                      (short)0, acc, false, false);
    }

    const int n = colBase + l15;
    const float bias = b_ig[n];
    const float wo   = W_out[n];

    float osum[8];
#pragma unroll
    for (int r = 0; r < 8; ++r) {
        int brow = rowBase + r + hi * 8;
        float ig  = fast_sigmoid(acc[r] + bias);
        float civ = ci[(long)brow * (LL * HH) + (long)t * HH + n];
        int cidx  = brow * HH + n;
        float cn  = c[cidx] + civ * ig;
        c[cidx]   = cn;
        hdst[cidx] = f32_to_bf16(cn);
        osum[r] = cn * wo;
    }

    // reduce the 16 partials of each row across the 16-lane group
#pragma unroll
    for (int r = 0; r < 8; ++r) {
        float s = osum[r];
        s += __shfl_xor(s, 1);
        s += __shfl_xor(s, 2);
        s += __shfl_xor(s, 4);
        s += __shfl_xor(s, 8);
        osum[r] = s;
    }
    if (l15 == 0) {
#pragma unroll
        for (int r = 0; r < 8; ++r) {
            int brow = rowBase + r + hi * 8;
            atomicAdd(&out[brow * LL + t], osum[r]);
        }
    }
}

// ---------------- launcher ----------------
extern "C" void kernel_launch(void* const* d_in, const int* in_sizes, int n_in,
                              void* d_out, int out_size, void* d_ws, size_t ws_size,
                              hipStream_t stream) {
    const float* obs   = (const float*)d_in[0];   // [128,512,144]
    const float* W_ci  = (const float*)d_in[1];   // [144,512]
    const float* b_ci  = (const float*)d_in[2];   // [512]
    const float* W_ig  = (const float*)d_in[3];   // [512,512]
    const float* b_ig  = (const float*)d_in[4];   // [512]
    const float* W_out = (const float*)d_in[5];   // [1,512]
    const float* b_out = (const float*)d_in[6];   // [1]
    float* out = (float*)d_out;                   // [128,512,1]

    char* ws = (char*)d_ws;
    size_t off = 0;
    unsigned short* Wt_ig  = (unsigned short*)(ws + off); off += (size_t)HH * HH * 2;      // 512 KB
    unsigned short* Wt_ci  = (unsigned short*)(ws + off); off += (size_t)HH * FP * 2;      // 160 KB
    unsigned short* obs_bf = (unsigned short*)(ws + off); off += (size_t)BB * LL * FP * 2; // 20 MB
    float*          ci     = (float*)(ws + off);          off += (size_t)BB * LL * HH * 4; // 134 MB
    float*          c      = (float*)(ws + off);          off += (size_t)BB * HH * 4;      // 256 KB
    unsigned short* hbuf   = (unsigned short*)(ws + off);                                  // 2x128 KB

    // prep
    conv_wig  <<<(HH * HH + 255) / 256, 256, 0, stream>>>(W_ig, Wt_ig);
    conv_wci  <<<(HH * FP + 255) / 256, 256, 0, stream>>>(W_ci, Wt_ci);
    conv_obs  <<<((long)BB * LL * FP + 255) / 256, 256, 0, stream>>>(obs, obs_bf);
    init_state<<<(BB * HH + 255) / 256, 256, 0, stream>>>(c, hbuf, out, b_out);

    // phase 1: ci GEMM — 4096*32 tiles, 8 waves per block
    ci_gemm<<<16384, 256, 0, stream>>>(obs_bf, Wt_ci, b_ci, ci);

    // phase 2: 512 serial steps (stream order = dependency order)
    for (int t = 0; t < LL; ++t) {
        const unsigned short* hs = hbuf + (size_t)(t & 1) * BB * HH;
        unsigned short*       hd = hbuf + (size_t)((t + 1) & 1) * BB * HH;
        step_kernel<<<64, 128, 0, stream>>>(hs, hd, Wt_ig, b_ig, ci, c, W_out, out, t);
    }
}
